// Transformer_19121194402067
// MI455X (gfx1250) — compile-verified, matrix operand order checked by reading, not run on hardware
//
#include <hip/hip_runtime.h>
#include <hip/hip_bf16.h>

// ---------------------------------------------------------------------------
// MI455X (gfx1250) implementation.
// All matmuls use V_WMMA_F32_16X16X32_F16 (f16 A/B, f32 accumulate).
// Dims: B=128, T=127, S=DK=128, H=8, L=6, P=8128.
// ---------------------------------------------------------------------------

typedef __attribute__((ext_vector_type(16))) _Float16 v16h;
typedef __attribute__((ext_vector_type(8)))  _Float16 v8h;
typedef __attribute__((ext_vector_type(8)))  float    v8f;

#define NB  128   // batch
#define NT  127   // raw tokens
#define NS  128   // tokens incl class token (== DK)
#define NDK 128
#define NH  8
#define NL  6
#define NP  8128  // (S*S-S)/2
#define NDI 48

// ---------------------------------------------------------------------------
// WMMA helpers.
// A (16x32 f16) fragment from a row-major matrix with leading dim 128:
//   lanes 0-15  : row = rowBase+lane,    K in {kBase+0..7} u {kBase+16..23}
//   lanes 16-31 : row = rowBase+lane-16, K in {kBase+8..15} u {kBase+24..31}
// B operand uses the mirrored layout, so the same loader works on an
// n-major (transposed) matrix: rows == output columns.
// ---------------------------------------------------------------------------
__device__ __forceinline__ v16h load_frag(const _Float16* p, int rowBase,
                                          int kBase, int lane) {
  const int half = (lane >> 4) & 1;
  const _Float16* q = p + (rowBase + (lane & 15)) * 128 + kBase + half * 8;
  v8h lo = *(const v8h*)(q);
  v8h hi = *(const v8h*)(q + 16);
  return __builtin_shufflevector(lo, hi, 0, 1, 2, 3, 4, 5, 6, 7,
                                 8, 9, 10, 11, 12, 13, 14, 15);
}

__device__ __forceinline__ v8f wmma_f16(v16h a, v16h b, v8f c) {
  return __builtin_amdgcn_wmma_f32_16x16x32_f16(false, a, false, b,
                                                (short)0, c, false, false);
}

// Row reductions inside one half-wave (a C-tile row lives in 16 lanes).
__device__ __forceinline__ float hredsum(float v) {
  v += __shfl_xor(v, 1, 16);
  v += __shfl_xor(v, 2, 16);
  v += __shfl_xor(v, 4, 16);
  v += __shfl_xor(v, 8, 16);
  return v;
}
__device__ __forceinline__ float hredmax(float v) {
  v = fmaxf(v, __shfl_xor(v, 1, 16));
  v = fmaxf(v, __shfl_xor(v, 2, 16));
  v = fmaxf(v, __shfl_xor(v, 4, 16));
  v = fmaxf(v, __shfl_xor(v, 8, 16));
  return v;
}

// ---------------------------------------------------------------------------
// 1) elementwise f32 -> f16 convert (weights)
// ---------------------------------------------------------------------------
extern "C" __global__ void __launch_bounds__(256)
k_cvt_f16(const float* __restrict__ src, _Float16* __restrict__ dst, int n) {
  int i = blockIdx.x * 256 + threadIdx.x;
  if (i < n) dst[i] = (_Float16)src[i];
}

// ---------------------------------------------------------------------------
// 2) cos/sin tables for all 48*8128 Givens angles
// ---------------------------------------------------------------------------
extern "C" __global__ void __launch_bounds__(256)
k_sincos(const float* __restrict__ phi, float* __restrict__ cs, int n) {
  int i = blockIdx.x * 256 + threadIdx.x;
  if (i < n) {
    float s, c;
    __sincosf(phi[i], &s, &c);
    cs[2 * i + 0] = c;
    cs[2 * i + 1] = s;
  }
}

// ---------------------------------------------------------------------------
// 3) Build orthogonal W per (l,h) via pyramid Givens scan.
// One block per matrix, thread c owns column c => no barriers in the scan.
// The per-k cascade is a register carry chain (1 fma dependent op / step).
// Output is W^T in f16 (n-major B operand for u @ W).
// Dynamic LDS: 128*128 f32 = 64KB.
// ---------------------------------------------------------------------------
extern "C" __global__ void __launch_bounds__(128)
k_build_w(const float* __restrict__ cs, _Float16* __restrict__ Wt16) {
  extern __shared__ float sW[];  // [row*128 + col]
  const int lh = blockIdx.x;
  const int c = threadIdx.x;

  for (int r = 0; r < 128; ++r) sW[r * 128 + c] = (r == c) ? 1.f : 0.f;

  const float* csb = cs + (size_t)lh * (2 * NP);
  int step = 0;
  for (int k = 0; k < NS - 1; ++k) {
    float carry = sW[(k + 1) * 128 + c];
    for (int i = k; i >= 0; --i, ++step) {
      float ct = csb[2 * step + 0];   // uniform -> scalar load
      float st = csb[2 * step + 1];
      float ra = sW[i * 128 + c];
      sW[(i + 1) * 128 + c] = st * ra + ct * carry;  // new W[i+1]
      carry = ct * ra - st * carry;                  // new W[i]
    }
    sW[c] = carry;  // row 0
  }
  __syncthreads();  // transpose readout touches other columns

  _Float16* out = Wt16 + (size_t)lh * (128 * 128);
  for (int idx = c; idx < 128 * 128; idx += 128) {
    int j = idx >> 7, d = idx & 127;
    out[idx] = (_Float16)sW[d * 128 + j];  // Wt[j][d] = W[d][j]
  }
}

// ---------------------------------------------------------------------------
// 4) Embedding + class token + norm0 layer norm. One block per (b,s).
// ---------------------------------------------------------------------------
extern "C" __global__ void __launch_bounds__(128)
k_embed(const float* __restrict__ x, const float* __restrict__ ew,
        const float* __restrict__ eb, const float* __restrict__ ct,
        const float* __restrict__ g, const float* __restrict__ bta,
        float* __restrict__ hout) {
  const int b = blockIdx.x >> 7;
  const int s = blockIdx.x & 127;
  const int d = threadIdx.x;
  __shared__ float xr[NDI];
  __shared__ float red[4], red2[4];

  float v;
  if (s == 0) {
    v = ct[d];
  } else {
    if (d < NDI) xr[d] = x[((size_t)b * NT + (s - 1)) * NDI + d];
    __syncthreads();
    float acc = eb[d];
    const float* wr = ew + d * NDI;
#pragma unroll 8
    for (int f = 0; f < NDI; ++f) acc += xr[f] * wr[f];
    v = acc;
  }

  float sum = v;
  for (int m = 1; m <= 16; m <<= 1) sum += __shfl_xor(sum, m, 32);
  if ((d & 31) == 0) red[d >> 5] = sum;
  __syncthreads();
  float mean = (red[0] + red[1] + red[2] + red[3]) * (1.f / 128.f);

  float dv = v - mean;
  float sq = dv * dv;
  for (int m = 1; m <= 16; m <<= 1) sq += __shfl_xor(sq, m, 32);
  if ((d & 31) == 0) red2[d >> 5] = sq;
  __syncthreads();
  float var = (red2[0] + red2[1] + red2[2] + red2[3]) * (1.f / 128.f);

  hout[(size_t)(b * 128 + s) * 128 + d] =
      dv * rsqrtf(var + 1e-5f) * g[d] + bta[d];
}

// ---------------------------------------------------------------------------
// 5) Per-layer prep: token norms + f16 packs of h and u=h/||h||.
// ---------------------------------------------------------------------------
extern "C" __global__ void __launch_bounds__(128)
k_prep(const float* __restrict__ hcur, _Float16* __restrict__ h16,
       _Float16* __restrict__ u16, float* __restrict__ nn) {
  const int row = blockIdx.x;  // b*128+s
  const int d = threadIdx.x;
  __shared__ float red[4];
  float v = hcur[(size_t)row * 128 + d];
  float sq = v * v;
  for (int m = 1; m <= 16; m <<= 1) sq += __shfl_xor(sq, m, 32);
  if ((d & 31) == 0) red[d >> 5] = sq;
  __syncthreads();
  float nrm = sqrtf(red[0] + red[1] + red[2] + red[3]);
  h16[(size_t)row * 128 + d] = (_Float16)v;
  u16[(size_t)row * 128 + d] = (_Float16)(v / nrm);
  if (d == 0) nn[row] = nrm;
}

// ---------------------------------------------------------------------------
// 6) h_next := merger_b[l] (accumulation target for atomic head merge)
// ---------------------------------------------------------------------------
extern "C" __global__ void __launch_bounds__(256)
k_fill(float* __restrict__ p, const float* __restrict__ mb, int l, int n) {
  int i = blockIdx.x * 256 + threadIdx.x;
  if (i < n) p[i] = mb[l];
}

// ---------------------------------------------------------------------------
// 7) V projection, stored transposed:  Vt[b,h][d][t] = (h[b] @ Vw^T + Vb)[t][d]
// Computed directly as Vt = Vw @ h^T : A = Vw16 rows (e x d), B = h16 rows.
// One block per (b,h); 8 waves; 32 WMMA / wave. Dynamic LDS: 32KB.
// ---------------------------------------------------------------------------
extern "C" __global__ void __launch_bounds__(256)
k_vproj(const _Float16* __restrict__ h16, const _Float16* __restrict__ Vw16,
        const float* __restrict__ Vb, _Float16* __restrict__ Vt, int l) {
  const int b = blockIdx.x >> 3;
  const int h = blockIdx.x & 7;
  extern __shared__ char smem[];
  _Float16* sH = (_Float16*)smem;  // h16[b], 128x128

  {
    const float4* src = (const float4*)(h16 + (size_t)b * 16384);
    float4* dst = (float4*)sH;
    for (int i = threadIdx.x; i < 2048; i += 256) dst[i] = src[i];
  }
  __syncthreads();

  const int lane = threadIdx.x & 31, w = threadIdx.x >> 5;
  const int half = lane >> 4, nlo = lane & 15, m0 = w * 16;

  const _Float16* Ap = Vw16 + (size_t)(l * NH + h) * 16384;
  v16h a[4];
#pragma unroll
  for (int ks = 0; ks < 4; ++ks) a[ks] = load_frag(Ap, m0, ks * 32, lane);

  const float* vb = Vb + (size_t)(l * NH + h) * 128;
  float vbr[8];
#pragma unroll
  for (int r = 0; r < 8; ++r) vbr[r] = vb[m0 + r + 8 * half];

  _Float16* out = Vt + (size_t)(b * NH + h) * 16384;
#pragma unroll
  for (int c = 0; c < 8; ++c) {
    v8f acc = {};
#pragma unroll
    for (int ks = 0; ks < 4; ++ks) {
      v16h bf = load_frag(sH, c * 16, ks * 32, lane);
      acc = wmma_f16(a[ks], bf, acc);
    }
#pragma unroll
    for (int r = 0; r < 8; ++r) {
      int e = m0 + r + 8 * half;
      int t = c * 16 + nlo;
      out[e * 128 + t] = (_Float16)(acc[r] + vbr[r]);
    }
  }
}

// ---------------------------------------------------------------------------
// 8) Fused head: T1 = u@W ; M = T1@u^T ; A = M^2*nn*nn*scale ; P = softmax ;
//    O = P@V ; LN(O + h) ; * merger_w ; atomic-add into h_next.
// One block per (b,h); 8 waves; each wave owns a 16-row slab; 96 WMMA / wave.
// Dynamic LDS: sU (u, later P) 32KB + sT (T1) 32KB = 64KB.
// ---------------------------------------------------------------------------
extern "C" __global__ void __launch_bounds__(256)
k_attn(const _Float16* __restrict__ u16, const _Float16* __restrict__ Wt16,
       const _Float16* __restrict__ Vt, const float* __restrict__ nn,
       const float* __restrict__ hcur, const float* __restrict__ ln_g,
       const float* __restrict__ ln_b, const float* __restrict__ merger_w,
       float* __restrict__ hnext, int l) {
  const int b = blockIdx.x >> 3;
  const int h = blockIdx.x & 7;
  extern __shared__ char smem[];
  _Float16* sU = (_Float16*)smem;            // u, later P
  _Float16* sT = (_Float16*)(smem + 32768);  // T1

  {
    const float4* src = (const float4*)(u16 + (size_t)b * 16384);
    float4* dst = (float4*)sU;
    for (int i = threadIdx.x; i < 2048; i += 256) dst[i] = src[i];
  }
  __syncthreads();

  const int lane = threadIdx.x & 31, w = threadIdx.x >> 5;
  const int half = lane >> 4, nlo = lane & 15, m0 = w * 16;

  // ---- GEMM1: T1 = u @ W (B streamed from L2-resident Wt) ----
  const _Float16* Wp = Wt16 + (size_t)(l * NH + h) * 16384;
  {
    v16h a[4];
#pragma unroll
    for (int ks = 0; ks < 4; ++ks) a[ks] = load_frag(sU, m0, ks * 32, lane);
#pragma unroll
    for (int c = 0; c < 8; ++c) {
      v8f acc = {};
#pragma unroll
      for (int ks = 0; ks < 4; ++ks) {
        v16h bf = load_frag(Wp, c * 16, ks * 32, lane);
        acc = wmma_f16(a[ks], bf, acc);
      }
#pragma unroll
      for (int r = 0; r < 8; ++r)
        sT[(m0 + r + 8 * half) * 128 + c * 16 + nlo] = (_Float16)acc[r];
    }
  }

  // ---- GEMM2: M = T1 @ u^T (wave reads only its own T1 rows) ----
  float mm[8][8];
  {
    v16h a[4];
#pragma unroll
    for (int ks = 0; ks < 4; ++ks) a[ks] = load_frag(sT, m0, ks * 32, lane);
#pragma unroll
    for (int c = 0; c < 8; ++c) {
      v8f acc = {};
#pragma unroll
      for (int ks = 0; ks < 4; ++ks) {
        v16h bf = load_frag(sU, c * 16, ks * 32, lane);
        acc = wmma_f16(a[ks], bf, acc);
      }
#pragma unroll
      for (int r = 0; r < 8; ++r) mm[c][r] = acc[r];
    }
  }

  // ---- scale + row softmax (rows live in one half-wave) ----
  const float* nnp = nn + (size_t)b * 128;
  float nrow[8], ncol[8];
#pragma unroll
  for (int r = 0; r < 8; ++r) nrow[r] = nnp[m0 + r + 8 * half];
#pragma unroll
  for (int c = 0; c < 8; ++c) ncol[c] = nnp[c * 16 + nlo];
  const float scale = 0.08838834764831845f;  // 1/sqrt(128)
#pragma unroll
  for (int c = 0; c < 8; ++c)
#pragma unroll
    for (int r = 0; r < 8; ++r) {
      float v = mm[c][r];
      mm[c][r] = v * v * nrow[r] * ncol[c] * scale;
    }
  float rinv[8];
#pragma unroll
  for (int r = 0; r < 8; ++r) {
    float mx = mm[0][r];
#pragma unroll
    for (int c = 1; c < 8; ++c) mx = fmaxf(mx, mm[c][r]);
    mx = hredmax(mx);
    float s = 0.f;
#pragma unroll
    for (int c = 0; c < 8; ++c) {
      float e = __expf(mm[c][r] - mx);
      mm[c][r] = e;
      s += e;
    }
    s = hredsum(s);
    rinv[r] = 1.f / s;
  }

  __syncthreads();  // all waves done reading sU; reuse it for P
#pragma unroll
  for (int c = 0; c < 8; ++c)
#pragma unroll
    for (int r = 0; r < 8; ++r)
      sU[(m0 + r + 8 * half) * 128 + c * 16 + nlo] =
          (_Float16)(mm[c][r] * rinv[r]);

  // ---- GEMM3: O = P @ V (B streamed from Vt, n-major) ----
  const _Float16* Vp = Vt + (size_t)(b * NH + h) * 16384;
  float oo[8][8];
  {
    v16h a[4];
#pragma unroll
    for (int ks = 0; ks < 4; ++ks) a[ks] = load_frag(sU, m0, ks * 32, lane);
#pragma unroll
    for (int c = 0; c < 8; ++c) {
      v8f acc = {};
#pragma unroll
      for (int ks = 0; ks < 4; ++ks) {
        v16h bf = load_frag(Vp, c * 16, ks * 32, lane);
        acc = wmma_f16(a[ks], bf, acc);
      }
#pragma unroll
      for (int r = 0; r < 8; ++r) oo[c][r] = acc[r];
    }
  }

  // ---- residual + layernorm + merger-weighted atomic accumulate ----
  const float* hr = hcur + (size_t)b * 16384;
  const float* gp = ln_g + (size_t)(l * NH + h) * 128;
  const float* bp = ln_b + (size_t)(l * NH + h) * 128;
  const float mw = merger_w[l * NH + h];
  float g8[8], b8[8];
#pragma unroll
  for (int c = 0; c < 8; ++c) {
    g8[c] = gp[c * 16 + nlo];
    b8[c] = bp[c * 16 + nlo];
  }
#pragma unroll
  for (int r = 0; r < 8; ++r) {
    const int s = m0 + r + 8 * half;
    float row[8], sum = 0.f, sq = 0.f;
#pragma unroll
    for (int c = 0; c < 8; ++c) {
      float v = oo[c][r] + hr[s * 128 + c * 16 + nlo];
      row[c] = v;
      sum += v;
      sq += v * v;
    }
    sum = hredsum(sum);
    sq = hredsum(sq);
    float mean = sum * (1.f / 128.f);
    float var = sq * (1.f / 128.f) - mean * mean;
    float rstd = rsqrtf(var + 1e-5f);
#pragma unroll
    for (int c = 0; c < 8; ++c) {
      float v = (row[c] - mean) * rstd * g8[c] + b8[c];
      atomicAdd(&hnext[(size_t)b * 16384 + s * 128 + c * 16 + nlo], v * mw);
    }
  }
}

// ---------------------------------------------------------------------------
// 9) Extract last token: out[b][d] = h[b][127][d]
// ---------------------------------------------------------------------------
extern "C" __global__ void __launch_bounds__(256)
k_tail(const float* __restrict__ hcur, float* __restrict__ out) {
  int i = blockIdx.x * 256 + threadIdx.x;
  if (i < NB * NDK) {
    int b = i >> 7, d = i & 127;
    out[i] = hcur[((size_t)b * 128 + 127) * 128 + d];
  }
}

// ---------------------------------------------------------------------------
// host orchestration
// ---------------------------------------------------------------------------
extern "C" void kernel_launch(void* const* d_in, const int* in_sizes, int n_in,
                              void* d_out, int out_size, void* d_ws,
                              size_t ws_size, hipStream_t stream) {
  (void)in_sizes; (void)n_in; (void)out_size; (void)ws_size;
  const float* x          = (const float*)d_in[0];
  const float* embed_w    = (const float*)d_in[1];
  const float* embed_b    = (const float*)d_in[2];
  const float* class_tok  = (const float*)d_in[3];
  const float* norm0_g    = (const float*)d_in[4];
  const float* norm0_b    = (const float*)d_in[5];
  const float* Vw         = (const float*)d_in[6];
  const float* Vb         = (const float*)d_in[7];
  const float* ln_g       = (const float*)d_in[8];
  const float* ln_b       = (const float*)d_in[9];
  const float* phi        = (const float*)d_in[10];
  const float* merger_w   = (const float*)d_in[11];
  const float* merger_b   = (const float*)d_in[12];
  float* out = (float*)d_out;

  // scratch carve-out (~62 MB total)
  char* ws = (char*)d_ws;
  size_t off = 0;
  auto carve = [&](size_t bytes) -> void* {
    off = (off + 255) & ~(size_t)255;
    void* p = ws + off;
    off += bytes;
    return p;
  };
  const size_t nTok = (size_t)NB * NS * NDK;        // 2M
  float*    hA   = (float*)carve(nTok * 4);          // 8 MB
  float*    hB   = (float*)carve(nTok * 4);          // 8 MB
  _Float16* h16  = (_Float16*)carve(nTok * 2);       // 4 MB
  _Float16* u16  = (_Float16*)carve(nTok * 2);       // 4 MB
  float*    nn   = (float*)carve((size_t)NB * NS * 4);
  _Float16* Vw16 = (_Float16*)carve((size_t)NL * NH * NDK * NDK * 2);
  _Float16* Wt16 = (_Float16*)carve((size_t)NL * NH * NS * NS * 2);
  float*    cs   = (float*)carve((size_t)NL * NH * NP * 2 * 4);
  _Float16* Vt   = (_Float16*)carve((size_t)NB * NH * NS * NDK * 2);  // 32 MB

  // one-time weight prep
  {
    int n = NL * NH * NDK * NDK;  // 786432
    k_cvt_f16<<<(n + 255) / 256, 256, 0, stream>>>(Vw, Vw16, n);
  }
  {
    int n = NL * NH * NP;  // 390144
    k_sincos<<<(n + 255) / 256, 256, 0, stream>>>(phi, cs, n);
  }
  k_build_w<<<NL * NH, 128, 128 * 128 * 4, stream>>>(cs, Wt16);

  // embed + norm0 -> hA
  k_embed<<<NB * NS, 128, 0, stream>>>(x, embed_w, embed_b, class_tok,
                                       norm0_g, norm0_b, hA);

  float* hcur = hA;
  float* hnext = hB;
  for (int l = 0; l < NL; ++l) {
    k_prep<<<NB * NS, 128, 0, stream>>>(hcur, h16, u16, nn);
    {
      int n = (int)nTok;
      k_fill<<<(n + 255) / 256, 256, 0, stream>>>(hnext, merger_b, l, n);
    }
    k_vproj<<<NB * NH, 256, 32768, stream>>>(h16, Vw16, Vb, Vt, l);
    k_attn<<<NB * NH, 256, 65536, stream>>>(u16, Wt16, Vt, nn, hcur, ln_g,
                                            ln_b, merger_w, hnext, l);
    float* t = hcur; hcur = hnext; hnext = t;
  }

  k_tail<<<(NB * NDK + 255) / 256, 256, 0, stream>>>(hcur, out);
}
